// Sketch_Segformer_43576738185512
// MI455X (gfx1250) — matrix operand order repaired and checked by reading, not compiled, not run on hardware
//
#include <hip/hip_runtime.h>

// ---------------------------------------------------------------------------
// Sketch-Segformer forward for MI455X (gfx1250, wave32, WMMA).
// Heavy GEMMs use v_wmma_f32_16x16x32_f16 (f16 in / f32 accumulate) with the
// weight panel staged into LDS (double-buffered) so B fragments are ds_load
// with immediate offsets instead of strided 64-bit-addressed global loads.
// Attention is fused flash-style per (batch, 16-row tile): scores + 1 (GA) or
// 8 (SA, stroke-masked) softmax rounds live in LDS; probs AND the V panel are
// stored as f16 in LDS and consumed directly by WMMA, so e[B,256,256] is never
// materialized and V is fetched from HBM exactly once per block.
// BatchNorm (training mode, global stats) = reduction kernel + fused apply.
// ---------------------------------------------------------------------------

typedef __attribute__((ext_vector_type(16))) _Float16 v16h;
typedef __attribute__((ext_vector_type(8)))  float    v8f;

#define NBATCH 128
#define NSEQ   256
#define NPTS   (NBATCH * NSEQ)   // 32768 points (M dimension of all GEMMs but the pooled one)

// ---------------- weight transpose + f32 -> f16 ----------------------------
// Wt[(dstOff+k)*Cout + o] = W[o*CinTot + srcOff + k]   (Cout is a power of 2)
__global__ void wconv_kernel(_Float16* __restrict__ Wt, const float* __restrict__ W,
                             int coutSh, int CinTot, int srcOff, int nTot, int dstOff)
{
    int idx = blockIdx.x * blockDim.x + threadIdx.x;
    if (idx >= nTot) return;
    int kk = idx >> coutSh, o = idx & ((1 << coutSh) - 1);
    Wt[(size_t)(dstOff + kk) << coutSh | o] =
        (_Float16)W[(size_t)o * CinTot + srcOff + kk];
}

// ---------------- WMMA GEMM: D[M,Cout] = A[M,K] * Wt[K,Cout] (+bias) --------
// A is f32 with row stride lda / column offset coA (so strided tensors such as
// the channel-concat buffers work without copies). Wt is f16 row-major [K,Cout].
// One wave computes one 16x16 f32 tile; block = 128 threads = 4 waves = 64 rows
// (all callers have M % 64 == 0, so control flow is block-uniform).
// A fragments: b128 loads + v_cvt_pk_f16_f32. B panel: cooperatively staged
// into double-buffered LDS (1 b64 global load + 1 ds_store_b64 per thread per
// K-step), fragments read back as ds_load_u16 with immediate offsets.
__global__ __launch_bounds__(128)
void gemm_wmma_kernel(const float* __restrict__ A, int lda, int coA,
                      const _Float16* __restrict__ Wt,
                      const float* __restrict__ bias,
                      float* __restrict__ D, int ldd, int coD,
                      int M, int K, int Cout)
{
    __shared__ _Float16 sB[2][32][16];   // 2 KB double-buffered weight panel

    const int t     = threadIdx.x;
    const int lane  = t & 31;
    const int wave  = t >> 5;            // 0..3
    const int row0  = (blockIdx.x * 4 + wave) * 16;
    const int col0  = blockIdx.y * 16;

    const int rA   = lane & 15;          // A row within tile
    const int hi   = lane >> 4;          // half-lane group
    const int k0A  = hi * 8;             // A: K 0..7 / 8..15 (+16 for a[8..15])
    const int k0B  = hi * 16;            // unused; B reads use A's k order (k0A)
    const int colB = lane & 15;

    // B staging: thread t stages panel row t/4, cols (t%4)*4 .. +3
    const int sRow = t >> 2;
    const int sCol = (t & 3) * 4;
    const _Float16* Wp = Wt + (size_t)sRow * Cout + col0 + sCol;
    const size_t wStep = (size_t)32 * Cout;

    v8f c = {0.f, 0.f, 0.f, 0.f, 0.f, 0.f, 0.f, 0.f};
    const float* Arow = A + (size_t)(row0 + rA) * lda + coA;

    int p = 0;
#pragma unroll 2
    for (int kk = 0; kk < K; kk += 32, Wp += wStep, p ^= 1) {
        if (kk + 32 < K) __builtin_prefetch(Arow + kk + 32, 0, 0);
        // ---- stage 32x16 B panel into LDS (alternating buffer) ----
        *(uint2*)(&sB[p][sRow][sCol]) = *(const uint2*)Wp;
        // ---- A fragment: 4 x b128 loads, then f32->f16 pack ----
        float av[16];
        *(float4*)(av + 0)  = *(const float4*)(Arow + kk + k0A);
        *(float4*)(av + 4)  = *(const float4*)(Arow + kk + k0A + 4);
        *(float4*)(av + 8)  = *(const float4*)(Arow + kk + k0A + 16);
        *(float4*)(av + 12) = *(const float4*)(Arow + kk + k0A + 20);
        v16h a, b;
#pragma unroll
        for (int i = 0; i < 16; ++i) a[i] = (_Float16)av[i];
        __syncthreads();
        // ---- B fragment from LDS: immediate-offset ds loads ----
#pragma unroll
        for (int i = 0; i < 16; ++i)
            b[i] = sB[p][k0A + i + (i >= 8 ? 8 : 0)][colB];

        c = __builtin_amdgcn_wmma_f32_16x16x32_f16(false, a, false, b,
                                                   (short)0, c, false, false);
    }

    const int orow = row0 + (hi << 3);
    const int ocol = col0 + (lane & 15);
    const float bv = bias ? bias[ocol] : 0.f;
#pragma unroll
    for (int j = 0; j < 8; ++j)
        D[(size_t)(orow + j) * ldd + coD + ocol] = c[j] + bv;
}

// ---------------- small conv (tiny Cin or Cout: 4->64, 64->8, 128->4) ------
// Cout is a power of two (coutSh = log2(Cout)).
__global__ void small_conv_kernel(float* __restrict__ D, int ldd, int coD,
                                  const float* __restrict__ X, int lda, int coA,
                                  const float* __restrict__ W,
                                  const float* __restrict__ bias,
                                  int nTot, int Cin, int coutSh)
{
    int idx = blockIdx.x * blockDim.x + threadIdx.x;
    if (idx >= nTot) return;
    int m = idx >> coutSh, o = idx & ((1 << coutSh) - 1);
    const float* xr = X + (size_t)m * lda + coA;
    const float* wr = W + (size_t)o * Cin;
    float acc = bias ? bias[o] : 0.f;
    for (int c = 0; c < Cin; ++c) acc += wr[c] * xr[c];
    D[(size_t)m * ldd + coD + o] = acc;
}

// ---------------- BatchNorm: per-channel stats over all M points -----------
__global__ void bn_stats_kernel(const float* __restrict__ X, int lda, int coA,
                                int M, float* __restrict__ mean, float* __restrict__ inv)
{
    __shared__ float s1[256], s2[256];
    const int c = blockIdx.x;
    float su = 0.f, sq = 0.f;
    for (int m = threadIdx.x; m < M; m += 256) {
        float v = X[(size_t)m * lda + coA + c];
        su += v; sq += v * v;
    }
    s1[threadIdx.x] = su; s2[threadIdx.x] = sq;
    __syncthreads();
    for (int s = 128; s > 0; s >>= 1) {
        if (threadIdx.x < s) {
            s1[threadIdx.x] += s1[threadIdx.x + s];
            s2[threadIdx.x] += s2[threadIdx.x + s];
        }
        __syncthreads();
    }
    if (threadIdx.x == 0) {
        float mu  = s1[0] / (float)M;
        float var = s2[0] / (float)M - mu * mu;
        mean[c] = mu;
        inv[c]  = rsqrtf(var + 1e-5f);
    }
}

// D = [relu]( g*(X-mean)*inv + b  [+ res] [+ D] ), C = power of two
__global__ void bn_apply_kernel(float* __restrict__ D, int ldd, int coD,
                                const float* __restrict__ X, int lda, int coA,
                                const float* __restrict__ res, int ldr, int coR,
                                const float* __restrict__ mean, const float* __restrict__ inv,
                                const float* __restrict__ g, const float* __restrict__ bt,
                                int nTot, int cSh, int relu, int accum)
{
    int idx = blockIdx.x * blockDim.x + threadIdx.x;
    if (idx >= nTot) return;
    int m = idx >> cSh, c = idx & ((1 << cSh) - 1);
    float y = g[c] * (X[(size_t)m * lda + coA + c] - mean[c]) * inv[c] + bt[c];
    if (res)   y += res[(size_t)m * ldr + coR + c];
    if (accum) y += D[(size_t)m * ldd + coD + c];
    if (relu)  y = fmaxf(y, 0.f);
    D[(size_t)m * ldd + coD + c] = y;
}

// ---------------- fused attention (GA: 1 round, SA: 8 masked rounds) -------
// Q,K: [M,8]   V: [M,32]   Out: [M,32]   grid = (NSEQ/16, NBATCH), block = 256.
// Scores for a 16-row tile are computed once into LDS; V for the whole batch
// is staged once into LDS as f16; every softmax round writes exp(l - rowmax)
// as f16 into LDS; waves 0/1 run the [16x256]x[256x32] product with a fully
// unrolled chain of v_wmma_f32_16x16x32_f16 reading A and B from LDS, folding
// 1/rowsum into the A fragment.
__global__ __launch_bounds__(256)
void attention_kernel(const float* __restrict__ Q, const float* __restrict__ Kc,
                      const float* __restrict__ V, const int* __restrict__ sidx,
                      float* __restrict__ Out, int masked)
{
    __shared__ float    sK[NSEQ][8];        // 8 KB
    __shared__ float    sQ[16][8];
    __shared__ float    sS[16][NSEQ];       // 16 KB raw scores (already /sqrt(8))
    __shared__ _Float16 sP[16][NSEQ];       // 8 KB  exp(l - max)
    __shared__ _Float16 sV[NSEQ][32];       // 16 KB V panel (f16), staged once
    __shared__ float    sRed[16][16];
    __shared__ float    sRow[16];
    __shared__ float    sRinv[16];
    __shared__ int      sIdx[NSEQ];

    const int b     = blockIdx.y;
    const int nBase = blockIdx.x * 16;
    const int t     = threadIdx.x;
    const int mBase = b * NSEQ;

#pragma unroll
    for (int c = 0; c < 8; ++c)
        sK[t][c] = Kc[(size_t)(mBase + t) * 8 + c];
#pragma unroll
    for (int c = 0; c < 32; ++c)
        sV[t][c] = (_Float16)V[(size_t)(mBase + t) * 32 + c];
    sIdx[t] = sidx[mBase + t];
    if (t < 16 * 8) {
        int r = t >> 3, c = t & 7;
        sQ[r][c] = Q[(size_t)(mBase + nBase + r) * 8 + c];
    }
    __syncthreads();

    const int r   = t >> 4;       // row 0..15 of the tile
    const int c00 = t & 15;       // this thread covers cols c00 + 16*j
    {
#pragma unroll
        for (int j = 0; j < 16; ++j) {
            int col = c00 + 16 * j;
            float acc = 0.f;
#pragma unroll
            for (int c = 0; c < 8; ++c) acc += sQ[r][c] * sK[col][c];
            sS[r][col] = acc * 0.35355339059327373f;   // 1/sqrt(8)
        }
    }
    __syncthreads();

    const int lane = t & 31;
    const int wave = t >> 5;
    v8f acc = {0.f, 0.f, 0.f, 0.f, 0.f, 0.f, 0.f, 0.f};

    const int nRounds = masked ? 8 : 1;
    for (int s = 0; s < nRounds; ++s) {
        // row max
        float mymax = -3.0e38f;
#pragma unroll
        for (int j = 0; j < 16; ++j) {
            int col = c00 + 16 * j;
            float l = sS[r][col];
            if (masked) l = (l + ((sIdx[col] == s) ? 0.f : 1e-6f)) * 10.f;
            mymax = fmaxf(mymax, l);
        }
        sRed[r][c00] = mymax;
        __syncthreads();
        if (c00 == 0) {
            float mx = sRed[r][0];
            for (int i = 1; i < 16; ++i) mx = fmaxf(mx, sRed[r][i]);
            sRow[r] = mx;
        }
        __syncthreads();

        // exp + row sum; stash exp() in f16 LDS for the WMMA A fragments
        const float rmax = sRow[r];
        float mysum = 0.f;
#pragma unroll
        for (int j = 0; j < 16; ++j) {
            int col = c00 + 16 * j;
            float l = sS[r][col];
            if (masked) l = (l + ((sIdx[col] == s) ? 0.f : 1e-6f)) * 10.f;
            float e = __expf(l - rmax);
            sP[r][col] = (_Float16)e;
            mysum += e;
        }
        sRed[r][c00] = mysum;
        __syncthreads();
        if (c00 == 0) {
            float tot = 0.f;
            for (int i = 0; i < 16; ++i) tot += sRed[r][i];
            sRinv[r] = 1.f / tot;
        }
        __syncthreads();

        // P[16,256] x V[256,32]: wave 0 -> out cols 0..15, wave 1 -> 16..31
        if (wave < 2) {
            const int hi   = lane >> 4;
            const int rA   = lane & 15;
            const int k0A  = hi * 8;
            const int k0B  = hi * 16;
            const int colV = wave * 16 + (lane & 15);
            const float rinv = sRinv[rA];
#pragma unroll
            for (int kk = 0; kk < NSEQ; kk += 32) {
                v16h aF, bF;
#pragma unroll
                for (int i = 0; i < 8; ++i) {
                    aF[i]     = (_Float16)((float)sP[rA][kk + k0A + i] * rinv);
                    aF[i + 8] = (_Float16)((float)sP[rA][kk + k0A + 16 + i] * rinv);
                }
#pragma unroll
                for (int i = 0; i < 16; ++i)
                    bF[i] = sV[kk + k0A + i + (i >= 8 ? 8 : 0)][colV];
                acc = __builtin_amdgcn_wmma_f32_16x16x32_f16(false, aF, false, bF,
                                                             (short)0, acc, false, false);
            }
        }
        __syncthreads();   // protect sP/sRed before next round
    }

    if (wave < 2) {
        const int hi   = lane >> 4;
        const int orow = nBase + hi * 8;
        const int ocol = wave * 16 + (lane & 15);
#pragma unroll
        for (int j = 0; j < 8; ++j)
            Out[(size_t)(mBase + orow + j) * 32 + ocol] = acc[j];
    }
}

// ---------------- misc elementwise helpers ---------------------------------
__global__ void add_pos_kernel(float* __restrict__ H, const float* __restrict__ pos,
                               int nTot, int cSh)
{
    int idx = blockIdx.x * blockDim.x + threadIdx.x;
    if (idx >= nTot) return;
    int m = idx >> cSh, c = idx & ((1 << cSh) - 1);
    H[(size_t)idx] += pos[((size_t)(m & (NSEQ - 1)) << cSh) + c];    // n = m % 256
}

__global__ void add_pooled_kernel(float* __restrict__ G, const float* __restrict__ P,
                                  int nTot, int cSh)
{
    int idx = blockIdx.x * blockDim.x + threadIdx.x;
    if (idx >= nTot) return;
    int m = idx >> cSh, c = idx & ((1 << cSh) - 1);
    G[(size_t)idx] += P[((size_t)(m >> 8) << cSh) + c];              // batch = m / 256
}

// fmax / favg over N per (batch, channel); pooled[b] = [fmax(1024) | favg(1024)]
__global__ __launch_bounds__(1024)
void pool_kernel(const float* __restrict__ F, float* __restrict__ pooled, int C)
{
    int b = blockIdx.x, c = threadIdx.x;
    float mx = -3.0e38f, su = 0.f;
    for (int n = 0; n < NSEQ; ++n) {
        float v = F[(size_t)(b * NSEQ + n) * C + c];
        mx = fmaxf(mx, v); su += v;
    }
    pooled[(size_t)b * 2 * C + c]     = mx;
    pooled[(size_t)b * 2 * C + C + c] = su * (1.f / (float)NSEQ);
}

__global__ void logsoftmax_kernel(float* __restrict__ out, const float* __restrict__ logits,
                                  int M, int C)
{
    int m = blockIdx.x * blockDim.x + threadIdx.x;
    if (m >= M) return;
    const float* r = logits + (size_t)m * C;
    float mx = r[0];
    for (int i = 1; i < C; ++i) mx = fmaxf(mx, r[i]);
    float s = 0.f;
    for (int i = 0; i < C; ++i) s += __expf(r[i] - mx);
    float lse = logf(s) + mx;
    for (int i = 0; i < C; ++i) out[(size_t)m * C + i] = r[i] - lse;
}

// ===========================================================================
// host side
// ===========================================================================
namespace {

inline int ilog2(int v) { int s = 0; while ((1 << s) < v) ++s; return s; }

inline void wconv(hipStream_t st, _Float16* Wt, const float* W,
                  int Cout, int CinTot, int srcOff, int nK, int dstOff)
{
    int tot = nK * Cout;
    wconv_kernel<<<(tot + 255) / 256, 256, 0, st>>>(Wt, W, ilog2(Cout), CinTot,
                                                    srcOff, tot, dstOff);
}

inline void gemm(hipStream_t st, const float* A, int lda, int coA,
                 const _Float16* Wt, const float* bias,
                 float* D, int ldd, int coD, int M, int K, int Cout)
{
    dim3 g(M / 64, Cout / 16);
    gemm_wmma_kernel<<<g, 128, 0, st>>>(A, lda, coA, Wt, bias, D, ldd, coD, M, K, Cout);
}

inline void sconv(hipStream_t st, float* D, int ldd, int coD,
                  const float* X, int lda, int coA,
                  const float* W, const float* bias, int M, int Cin, int Cout)
{
    int tot = M * Cout;
    small_conv_kernel<<<(tot + 255) / 256, 256, 0, st>>>(D, ldd, coD, X, lda, coA,
                                                         W, bias, tot, Cin, ilog2(Cout));
}

inline void bn(hipStream_t st, float* D, int ldd, int coD,
               const float* X, int lda, int coA,
               const float* res, int ldr, int coR,
               const float* g, const float* bt, float* mean, float* inv,
               int M, int C, int relu, int accum)
{
    bn_stats_kernel<<<C, 256, 0, st>>>(X, lda, coA, M, mean, inv);
    int tot = M * C;
    bn_apply_kernel<<<(tot + 255) / 256, 256, 0, st>>>(D, ldd, coD, X, lda, coA,
                                                       res, ldr, coR, mean, inv,
                                                       g, bt, tot, ilog2(C), relu, accum);
}

struct DualWt { _Float16 *vga, *lbga, *vsa, *lbsa, *w1, *w2; };
struct Scratch {
    float *qb, *kb, *q2b, *k2b, *vb, *att, *lin, *fbuf, *t64b, *mlph, *mean, *inv;
};

// d_in layout: setup_inputs() recursive dict (insertion) order.
// 0:x 1:stroke_idx 2:pos_emb 3..8:mlp1 9..32:db1 33..56:db2
// 57..59:lbr1 60..62:lbr2 63..65:mlp2_w1 66:mlp2_w2
inline const float* P(void* const* din, int i) { return (const float*)din[i]; }

void run_dual_block(hipStream_t st, void* const* din, int pb,
                    const float* X, int ldx, int cox,
                    float* outF, int ldo, int coo,
                    const DualWt& wt, const Scratch& S, const int* sidx)
{
    const int M = NPTS;
    dim3 agrid(NSEQ / 16, NBATCH);

    // ---- GA branch: f_ga = bn(attn) + bn(lin) -> fbuf cols 0..31
    sconv(st, S.qb, 8, 0, X, ldx, cox, P(din, pb + 0), nullptr, M, 64, 8);
    sconv(st, S.kb, 8, 0, X, ldx, cox, P(din, pb + 1), nullptr, M, 64, 8);
    gemm(st, X, ldx, cox, wt.vga, P(din, pb + 3), S.vb, 32, 0, M, 64, 32);
    attention_kernel<<<agrid, 256, 0, st>>>(S.qb, S.kb, S.vb, sidx, S.att, 0);
    bn(st, S.fbuf, 64, 0, S.att, 32, 0, nullptr, 0, 0,
       P(din, pb + 7), P(din, pb + 8), S.mean, S.inv, M, 32, 0, 0);
    gemm(st, X, ldx, cox, wt.lbga, nullptr, S.lin, 32, 0, M, 64, 32);
    bn(st, S.fbuf, 64, 0, S.lin, 32, 0, nullptr, 0, 0,
       P(din, pb + 5), P(din, pb + 6), S.mean, S.inv, M, 32, 0, 1);

    // ---- SA branch: 8 stroke-masked rounds -> fbuf cols 32..63
    sconv(st, S.q2b, 8, 0, X, ldx, cox, P(din, pb + 9),  nullptr, M, 64, 8);
    sconv(st, S.k2b, 8, 0, X, ldx, cox, P(din, pb + 10), nullptr, M, 64, 8);
    gemm(st, X, ldx, cox, wt.vsa, P(din, pb + 12), S.vb, 32, 0, M, 64, 32);
    attention_kernel<<<agrid, 256, 0, st>>>(S.q2b, S.k2b, S.vb, sidx, S.att, 1);
    bn(st, S.fbuf, 64, 32, S.att, 32, 0, nullptr, 0, 0,
       P(din, pb + 16), P(din, pb + 17), S.mean, S.inv, M, 32, 0, 0);
    gemm(st, X, ldx, cox, wt.lbsa, nullptr, S.lin, 32, 0, M, 64, 32);
    bn(st, S.fbuf, 64, 32, S.lin, 32, 0, nullptr, 0, 0,
       P(din, pb + 14), P(din, pb + 15), S.mean, S.inv, M, 32, 0, 1);

    // ---- MLP + residual: out = relu(bn(W2 * relu(bn(W1 * f))) + f)
    gemm(st, S.fbuf, 64, 0, wt.w1, nullptr, S.mlph, 256, 0, M, 64, 256);
    bn(st, S.mlph, 256, 0, S.mlph, 256, 0, nullptr, 0, 0,
       P(din, pb + 19), P(din, pb + 20), S.mean, S.inv, M, 256, 1, 0);
    gemm(st, S.mlph, 256, 0, wt.w2, nullptr, S.t64b, 64, 0, M, 256, 64);
    bn(st, outF, ldo, coo, S.t64b, 64, 0, S.fbuf, 64, 0,
       P(din, pb + 22), P(din, pb + 23), S.mean, S.inv, M, 64, 1, 0);
}

} // namespace

extern "C" void kernel_launch(void* const* d_in, const int* in_sizes, int n_in,
                              void* d_out, int out_size, void* d_ws, size_t ws_size,
                              hipStream_t stream)
{
    (void)in_sizes; (void)n_in; (void)out_size; (void)ws_size;
    const int M = NPTS;
    const float* x    = P(d_in, 0);
    const int*   sidx = (const int*)d_in[1];

    // ---- workspace carve-up (~250 MB total) ----
    char*  base = (char*)d_ws;
    size_t cur  = 0;
    auto alloc = [&](size_t bytes) -> void* {
        void* p = base + cur;
        cur = (cur + bytes + 255) & ~(size_t)255;
        return p;
    };
    float* t64a  = (float*)alloc((size_t)M * 64 * 4);
    float* h0    = (float*)alloc((size_t)M * 64 * 4);
    float* fcat  = (float*)alloc((size_t)M * 256 * 4);   // x1..x4 channel-concat; later h128
    float* f1024 = (float*)alloc((size_t)M * 1024 * 4);
    float* mlph  = (float*)alloc((size_t)M * 256 * 4);   // dual-block hidden; later f256
    Scratch S;
    S.qb   = (float*)alloc((size_t)M * 8 * 4);
    S.kb   = (float*)alloc((size_t)M * 8 * 4);
    S.q2b  = (float*)alloc((size_t)M * 8 * 4);
    S.k2b  = (float*)alloc((size_t)M * 8 * 4);
    S.vb   = (float*)alloc((size_t)M * 32 * 4);
    S.att  = (float*)alloc((size_t)M * 32 * 4);
    S.lin  = (float*)alloc((size_t)M * 32 * 4);
    S.fbuf = (float*)alloc((size_t)M * 64 * 4);
    S.t64b = (float*)alloc((size_t)M * 64 * 4);
    S.mlph = mlph;
    S.mean = (float*)alloc(1024 * 4);
    S.inv  = (float*)alloc(1024 * 4);
    float* pooled = (float*)alloc((size_t)NBATCH * 2048 * 4);
    float* pout   = (float*)alloc((size_t)NBATCH * 256 * 4);
    float* logits = (float*)alloc((size_t)M * 4 * 4);

    _Float16* wt_m1w2 = (_Float16*)alloc((size_t)64 * 64 * 2);
    DualWt dwt[2];
    for (int j = 0; j < 2; ++j) {
        dwt[j].vga  = (_Float16*)alloc((size_t)64 * 32 * 2);
        dwt[j].lbga = (_Float16*)alloc((size_t)64 * 32 * 2);
        dwt[j].vsa  = (_Float16*)alloc((size_t)64 * 32 * 2);
        dwt[j].lbsa = (_Float16*)alloc((size_t)64 * 32 * 2);
        dwt[j].w1   = (_Float16*)alloc((size_t)64 * 256 * 2);
        dwt[j].w2   = (_Float16*)alloc((size_t)256 * 64 * 2);
    }
    _Float16* wt_lbr1 = (_Float16*)alloc((size_t)256 * 1024 * 2);
    _Float16* wt_l2b  = (_Float16*)alloc((size_t)1024 * 256 * 2);
    _Float16* wt_l2ac = (_Float16*)alloc((size_t)2048 * 256 * 2);
    _Float16* wt_m2w1 = (_Float16*)alloc((size_t)256 * 128 * 2);

    // ---- convert GEMM weights to transposed f16 once per launch ----
    wconv(stream, wt_m1w2, P(d_in, 6), 64, 64, 0, 64, 0);
    const int pbase[2] = {9, 33};                      // db1, db2
    for (int j = 0; j < 2; ++j) {
        int pb = pbase[j];
        wconv(stream, dwt[j].vga,  P(d_in, pb + 2),  32, 64, 0, 64, 0);
        wconv(stream, dwt[j].lbga, P(d_in, pb + 4),  32, 64, 0, 64, 0);
        wconv(stream, dwt[j].vsa,  P(d_in, pb + 11), 32, 64, 0, 64, 0);
        wconv(stream, dwt[j].lbsa, P(d_in, pb + 13), 32, 64, 0, 64, 0);
        wconv(stream, dwt[j].w1,   P(d_in, pb + 18), 256, 64, 0, 64, 0);
        wconv(stream, dwt[j].w2,   P(d_in, pb + 21), 64, 256, 0, 256, 0);
    }
    wconv(stream, wt_lbr1, P(d_in, 57), 1024, 256, 0, 256, 0);
    // lbr2_w [256,3072]: cols 0..1023 act on fmax, 1024..2047 on f, 2048..3071 on favg.
    wconv(stream, wt_l2b,  P(d_in, 60), 256, 3072, 1024, 1024, 0);
    wconv(stream, wt_l2ac, P(d_in, 60), 256, 3072, 0,    1024, 0);
    wconv(stream, wt_l2ac, P(d_in, 60), 256, 3072, 2048, 1024, 1024);
    wconv(stream, wt_m2w1, P(d_in, 63), 128, 256, 0, 256, 0);

    // ---- stem: mlp1 (4->64->64) + positional embedding ----
    sconv(stream, t64a, 64, 0, x, 4, 0, P(d_in, 3), nullptr, M, 4, 64);
    bn(stream, t64a, 64, 0, t64a, 64, 0, nullptr, 0, 0,
       P(d_in, 4), P(d_in, 5), S.mean, S.inv, M, 64, 1, 0);
    gemm(stream, t64a, 64, 0, wt_m1w2, nullptr, h0, 64, 0, M, 64, 64);
    bn(stream, h0, 64, 0, h0, 64, 0, nullptr, 0, 0,
       P(d_in, 7), P(d_in, 8), S.mean, S.inv, M, 64, 1, 0);
    add_pos_kernel<<<(M * 64 + 255) / 256, 256, 0, stream>>>(h0, P(d_in, 2), M * 64, 6);

    // ---- dual blocks (db1 once, db2 reused 3x); outputs land in fcat ----
    run_dual_block(stream, d_in, 9,  h0,   64,  0,   fcat, 256, 0,   dwt[0], S, sidx);
    run_dual_block(stream, d_in, 33, fcat, 256, 0,   fcat, 256, 64,  dwt[1], S, sidx);
    run_dual_block(stream, d_in, 33, fcat, 256, 64,  fcat, 256, 128, dwt[1], S, sidx);
    run_dual_block(stream, d_in, 33, fcat, 256, 128, fcat, 256, 192, dwt[1], S, sidx);

    // ---- lbr1: 256 -> 1024, BN + relu ----
    gemm(stream, fcat, 256, 0, wt_lbr1, nullptr, f1024, 1024, 0, M, 256, 1024);
    bn(stream, f1024, 1024, 0, f1024, 1024, 0, nullptr, 0, 0,
       P(d_in, 58), P(d_in, 59), S.mean, S.inv, M, 1024, 1, 0);

    // ---- pooled (fmax|favg) part of lbr2 as a [128,2048]x[2048,256] GEMM ----
    pool_kernel<<<NBATCH, 1024, 0, stream>>>(f1024, pooled, 1024);
    gemm(stream, pooled, 2048, 0, wt_l2ac, nullptr, pout, 256, 0, NBATCH, 2048, 256);

    // ---- lbr2 main: f1024 -> 256, + pooled contribution, BN + relu ----
    gemm(stream, f1024, 1024, 0, wt_l2b, nullptr, mlph, 256, 0, M, 1024, 256);
    add_pooled_kernel<<<(M * 256 + 255) / 256, 256, 0, stream>>>(mlph, pout, M * 256, 8);
    bn(stream, mlph, 256, 0, mlph, 256, 0, nullptr, 0, 0,
       P(d_in, 61), P(d_in, 62), S.mean, S.inv, M, 256, 1, 0);

    // ---- mlp2: 256 -> 128 (BN+relu) -> 4 -> log_softmax ----
    float* h128 = fcat;   // reuse
    gemm(stream, mlph, 256, 0, wt_m2w1, nullptr, h128, 128, 0, M, 256, 128);
    bn(stream, h128, 128, 0, h128, 128, 0, nullptr, 0, 0,
       P(d_in, 64), P(d_in, 65), S.mean, S.inv, M, 128, 1, 0);
    sconv(stream, logits, 4, 0, h128, 128, 0, P(d_in, 66), nullptr, M, 128, 4);
    logsoftmax_kernel<<<(M + 255) / 256, 256, 0, stream>>>((float*)d_out, logits, M, 4);
}
